// PreNormChannelVideoAttention_8486855376945
// MI455X (gfx1250) — matrix operand order repaired and checked by reading, not compiled, not run on hardware
//
#include <hip/hip_runtime.h>
#include <hip/hip_bf16.h>

typedef __attribute__((ext_vector_type(16))) __bf16 v16bf;
typedef __attribute__((ext_vector_type(8)))  float  v8f;

#define C_DIM 1024
#define P_DIM 1024   // H*W
#define T_DIM 16
#define B_DIM 2
#define EPSV  1e-5f

// ---------------------------------------------------------------------------
// Kernel 1: per-pixel LayerNorm stats (mean, rstd) over C.
// lanes = consecutive pixels (w) -> coalesced 128B loads per channel.
// ---------------------------------------------------------------------------
__global__ void ln_stats_kernel(const float* __restrict__ x, float2* __restrict__ stats) {
    int idx = blockIdx.x * 256 + threadIdx.x;      // 0 .. 32767 (bt*1024 + p)
    int bt = idx >> 10;
    int p  = idx & 1023;
    const float* px = x + (size_t)bt * C_DIM * P_DIM + p;
    float s = 0.f, ss = 0.f;
#pragma unroll 4
    for (int c = 0; c < C_DIM; ++c) {
        float v = px[(size_t)c * P_DIM];
        s += v; ss += v * v;
    }
    float mu  = s * (1.f / C_DIM);
    float var = ss * (1.f / C_DIM) - mu * mu;
    stats[idx] = make_float2(mu, rsqrtf(var + EPSV));
}

// ---------------------------------------------------------------------------
// Kernel 2: normalize + transpose to bf16 A[b,t,q,c] with q = w'*32 + h'
// (w-major pixel order to match the reference's flatten, so final output
// stores are contiguous). 32x32 LDS tile transpose; both global phases
// coalesced.
// ---------------------------------------------------------------------------
__global__ void norm_transpose_kernel(const float* __restrict__ x,
                                      const float2* __restrict__ stats,
                                      const float* __restrict__ lnw,
                                      const float* __restrict__ lnb,
                                      __bf16* __restrict__ A) {
    __shared__ float tile[32][33];
    int bt = blockIdx.y;            // b*16 + t
    int h0 = blockIdx.x;            // pixel row: p = h0*32 + w'
    int tx = threadIdx.x & 31;
    int ty = threadIdx.x >> 5;      // 0..7
    int p  = h0 * 32 + tx;          // natural pixel for the read phase
    float2 st = stats[bt * P_DIM + p];
    const float* px = x + (size_t)bt * C_DIM * P_DIM;

    for (int c0 = 0; c0 < C_DIM; c0 += 32) {
#pragma unroll
        for (int r = 0; r < 4; ++r) {
            int cl = ty + r * 8;
            int c  = c0 + cl;
            float v = px[(size_t)c * P_DIM + p];
            tile[cl][tx] = (v - st.x) * st.y * lnw[c] + lnb[c];
        }
        __syncthreads();
#pragma unroll
        for (int r = 0; r < 4; ++r) {
            int pl = ty + r * 8;            // w' = pl, h' = h0
            int q  = pl * 32 + h0;          // w-major row index
            A[((size_t)bt * P_DIM + q) * C_DIM + c0 + tx] = (__bf16)tile[tx][pl];
        }
        __syncthreads();
    }
}

// ---------------------------------------------------------------------------
// Kernel 3: pooled[b,t,c] = mean over q of A (bf16 -> f32 accumulate).
// ---------------------------------------------------------------------------
__global__ void pooled_kernel(const __bf16* __restrict__ A, float* __restrict__ pooled) {
    int bt  = blockIdx.x;
    int tid = threadIdx.x;
    const __bf16* base = A + (size_t)bt * P_DIM * C_DIM;
    float acc[4] = {0.f, 0.f, 0.f, 0.f};
    for (int q = 0; q < P_DIM; ++q) {
#pragma unroll
        for (int k = 0; k < 4; ++k)
            acc[k] += (float)base[(size_t)q * C_DIM + k * 256 + tid];
    }
#pragma unroll
    for (int k = 0; k < 4; ++k)
        pooled[bt * C_DIM + k * 256 + tid] = acc[k] * (1.f / P_DIM);
}

// ---------------------------------------------------------------------------
// Kernel 4: cast Wv -> bf16 (row-major [c'][c] == WMMA B-fragment layout).
// ---------------------------------------------------------------------------
__global__ void wv_bf16_kernel(const float* __restrict__ Wv, __bf16* __restrict__ Wvb) {
    int i = blockIdx.x * 256 + threadIdx.x;
    Wvb[i] = (__bf16)Wv[i];
}

// ---------------------------------------------------------------------------
// Kernel 5: Q/K projections + scaled scores + softmax per (b, head).
// ---------------------------------------------------------------------------
__global__ void qk_attn_kernel(const float* __restrict__ pooled,
                               const float* __restrict__ Wq,
                               const float* __restrict__ Wk,
                               float* __restrict__ attn) {
    __shared__ float qs[16][64];
    __shared__ float ks[16][64];
    int head = blockIdx.x, b = blockIdx.y;
    int tid = threadIdx.x;
    int d = tid & 63, sg = tid >> 6;                 // sg 0..3
    const float* wqr = Wq + (size_t)(head * 64 + d) * C_DIM;
    const float* wkr = Wk + (size_t)(head * 64 + d) * C_DIM;
    const float* pb  = pooled + b * T_DIM * C_DIM;
    float aq[4] = {0.f, 0.f, 0.f, 0.f};
    float ak[4] = {0.f, 0.f, 0.f, 0.f};
    for (int c = 0; c < C_DIM; ++c) {
        float wq = wqr[c], wk = wkr[c];
#pragma unroll
        for (int s4 = 0; s4 < 4; ++s4) {
            float pv = pb[(sg + s4 * 4) * C_DIM + c];
            aq[s4] += pv * wq;
            ak[s4] += pv * wk;
        }
    }
#pragma unroll
    for (int s4 = 0; s4 < 4; ++s4) {
        qs[sg + s4 * 4][d] = aq[s4];
        ks[sg + s4 * 4][d] = ak[s4];
    }
    __syncthreads();
    int s = tid >> 4, t = tid & 15;
    float dot = 0.f;
#pragma unroll
    for (int dd = 0; dd < 64; ++dd) dot += qs[s][dd] * ks[t][dd];
    dot *= 0.125f;                                   // 1/sqrt(dh=64)
    float mx = dot;
#pragma unroll
    for (int o = 8; o >= 1; o >>= 1) mx = fmaxf(mx, __shfl_xor(mx, o, 16));
    float e = __expf(dot - mx);
    float sum = e;
#pragma unroll
    for (int o = 8; o >= 1; o >>= 1) sum += __shfl_xor(sum, o, 16);
    attn[((b * 16 + head) * 16 + s) * 16 + t] = e / sum;
}

// ---------------------------------------------------------------------------
// Kernel 6: V = A @ Wv^T via v_wmma_f32_16x16x32_bf16, fused attention
// epilogue (mix over t in LDS) + direct store to [B,T,C,H,W] output.
// Grid: (64 q-tiles, 16 heads, 2 b). 256 threads = 8 waves.
// Wave w owns t = {2w, 2w+1} x 4 c'-tiles -> 8 f32 16x16 accumulators.
// ---------------------------------------------------------------------------
__global__ void __launch_bounds__(256)
gemm_attn_kernel(const __bf16* __restrict__ A, const __bf16* __restrict__ Wv,
                 const float* __restrict__ attn, float* __restrict__ out) {
    __shared__ float v_lds[16][16][65];   // [t][p][c'] padded (bank-conflict free)
    __shared__ float attn_s[16][16];

    const int q0   = blockIdx.x * 16;
    const int head = blockIdx.y;
    const int c0   = head * 64;
    const int b    = blockIdx.z;
    const int tid  = threadIdx.x;
    const int lane = tid & 31;
    const int wid  = tid >> 5;            // 0..7
    const int m    = lane & 15;           // row-in-tile
    const int hk   = lane >> 4;           // K-half selector

    attn_s[tid >> 4][tid & 15] = attn[(b * 16 + head) * 256 + tid];

    v8f acc[2][4] = {};

    for (int k0 = 0; k0 < C_DIM; k0 += 32) {
        v16bf a[2], bb[4];
#pragma unroll
        for (int rt = 0; rt < 2; ++rt) {
            int t = wid * 2 + rt;
            const __bf16* pA = A + (((size_t)(b * 16 + t) * P_DIM) + q0 + m) * C_DIM
                                 + k0 + hk * 8;
            uint4* dst = (uint4*)&a[rt];
            dst[0] = *(const uint4*)(pA);        // K chunk lo (16B)
            dst[1] = *(const uint4*)(pA + 16);   // K chunk hi (16B, +16 elems)
        }
#pragma unroll
        for (int j = 0; j < 4; ++j) {
            const __bf16* pB = Wv + (size_t)(c0 + j * 16 + m) * C_DIM + k0 + hk * 8;
            uint4* dst = (uint4*)&bb[j];
            dst[0] = *(const uint4*)(pB);        // K chunk lo (16B)
            dst[1] = *(const uint4*)(pB + 16);   // K chunk hi (16B, +16 elems)
        }
#pragma unroll
        for (int rt = 0; rt < 2; ++rt)
#pragma unroll
            for (int j = 0; j < 4; ++j)
                acc[rt][j] = __builtin_amdgcn_wmma_f32_16x16x32_bf16(
                    false, a[rt], false, bb[j], (short)0, acc[rt][j], false, false);
    }

    // Stage V tiles in LDS: lane L holds col N=L%16, VGPR i -> row M=i+8*(L/16)
#pragma unroll
    for (int rt = 0; rt < 2; ++rt) {
        int t = wid * 2 + rt;
#pragma unroll
        for (int j = 0; j < 4; ++j)
#pragma unroll
            for (int i = 0; i < 8; ++i)
                v_lds[t][i + 8 * hk][j * 16 + m] = acc[rt][j][i];
    }
    __syncthreads();

    // Attention mix over t; stores contiguous in q.
    int p  = tid & 15;
    int cg = tid >> 4;                    // 0..15
#pragma unroll
    for (int q4 = 0; q4 < 4; ++q4) {
        int cc = q4 * 16 + cg;
        float vt[16];
#pragma unroll
        for (int t = 0; t < 16; ++t) vt[t] = v_lds[t][p][cc];
#pragma unroll
        for (int s = 0; s < 16; ++s) {
            float o = 0.f;
#pragma unroll
            for (int t = 0; t < 16; ++t) o += attn_s[s][t] * vt[t];
            out[((size_t)(b * 16 + s) * C_DIM + c0 + cc) * P_DIM + q0 + p] = o;
        }
    }
}

// ---------------------------------------------------------------------------
extern "C" void kernel_launch(void* const* d_in, const int* in_sizes, int n_in,
                              void* d_out, int out_size, void* d_ws, size_t ws_size,
                              hipStream_t stream) {
    (void)in_sizes; (void)n_in; (void)out_size; (void)ws_size;
    const float* x   = (const float*)d_in[0];
    const float* lnw = (const float*)d_in[1];
    const float* lnb = (const float*)d_in[2];
    const float* Wq  = (const float*)d_in[3];
    const float* Wk  = (const float*)d_in[4];
    const float* Wv  = (const float*)d_in[5];
    float* out = (float*)d_out;

    char* ws = (char*)d_ws;
    float2* stats  = (float2*)(ws);                  // 32768 * 8  = 256 KB
    float*  pooled = (float*)(ws + 262144);          // 32768 * 4  = 128 KB
    float*  attnb  = (float*)(ws + 393216);          // 8192  * 4  =  32 KB
    __bf16* Wvb    = (__bf16*)(ws + 458752);         // 1M    * 2  =   2 MB
    __bf16* A      = (__bf16*)(ws + 2621440);        // 33.5M * 2  =  64 MB

    ln_stats_kernel<<<128, 256, 0, stream>>>(x, stats);
    norm_transpose_kernel<<<dim3(32, 32), 256, 0, stream>>>(x, stats, lnw, lnb, A);
    wv_bf16_kernel<<<4096, 256, 0, stream>>>(Wv, Wvb);
    pooled_kernel<<<32, 256, 0, stream>>>(A, pooled);
    qk_attn_kernel<<<dim3(16, 2), 256, 0, stream>>>(pooled, Wq, Wk, attnb);
    gemm_attn_kernel<<<dim3(64, 16, 2), 256, 0, stream>>>(A, Wvb, attnb, out);
}
